// NodeUpdateNet_43112881717683
// MI455X (gfx1250) — compile-verified, matrix operand order checked by reading, not run on hardware
//
#include <hip/hip_runtime.h>
#include <hip/hip_bf16.h>
#include <stdint.h>

// ---------------------------------------------------------------------------
// MI455X (gfx1250) implementation.
//   - bf16 WMMA (v_wmma_f32_16x16x32_bf16) for all GEMMs, f32 accumulate
//   - weights pre-swizzled to the WMMA B-operand VGPR layout (bf16, in d_ws)
//   - segment-sum via global_atomic_add_f32 (inline CDNA5 asm)
//   - per-wave 16-row tiles; LayerNorm via LDS + wave32 xor-shuffle
// ---------------------------------------------------------------------------

#define XDIM 128
#define EDIM 128
#define HID  256
#define LN_EPS 1e-5f

typedef __attribute__((ext_vector_type(16))) __bf16 v16bf;
typedef __attribute__((ext_vector_type(8)))  float  v8f;

struct Q2 { uint4 a, b; };  // 32 bytes == 16 bf16

static __device__ __forceinline__ unsigned short bfbits(float f) {
    unsigned int u = __builtin_bit_cast(unsigned int, f);
    u = (u + 0x7FFFu + ((u >> 16) & 1u)) >> 16;   // round-to-nearest-even
    return (unsigned short)u;
}
static __device__ __forceinline__ unsigned int pk2(float a, float b) {
    return (unsigned int)bfbits(a) | ((unsigned int)bfbits(b) << 16);
}
static __device__ __forceinline__ v16bf ldA16(const __bf16* p0, const __bf16* p1) {
    Q2 q;
    q.a = *(const uint4*)p0;
    q.b = *(const uint4*)p1;
    return __builtin_bit_cast(v16bf, q);
}
static __device__ __forceinline__ v16bf ldB16(const __bf16* wmat, int tile, int lane) {
    const uint4* tp = (const uint4*)wmat + ((size_t)tile * 64 + lane * 2);
    Q2 q;
    q.a = tp[0];
    q.b = tp[1];
    return __builtin_bit_cast(v16bf, q);
}
// GLOBAL_ATOMIC_ADD_F32 (ISA 15.18.3 op 86), no return -> STOREcnt path
static __device__ __forceinline__ void gatomic_add(float* p, float v) {
    asm volatile("global_atomic_add_f32 %0, %1, off"
                 :
                 : "v"((unsigned long long)(uintptr_t)p), "v"(v)
                 : "memory");
}

struct MlpP {
    const __bf16* w0; const float* b0; const float* g0; const float* bb0;
    const __bf16* w1; const float* b1; const float* g1; const float* bb1;
};
struct EdgeParams { MlpP mlp[2]; };        // [0]=d2t (row<col), [1]=t2d (row>col)
struct NodeParams { MlpP node; MlpP self; };

// ---------------------------------------------------------------------------
// workspace zeroing
__global__ __launch_bounds__(256) void zero_kernel(float4* p, int n4) {
    int i = blockIdx.x * blockDim.x + threadIdx.x;
    if (i < n4) p[i] = make_float4(0.f, 0.f, 0.f, 0.f);
}

// ---------------------------------------------------------------------------
// fp32 W[dout][din] (row-major) -> bf16 tiles in WMMA-B lane layout.
// Tile t = nt*KC + kc holds outputs nt*16..+15 x inputs kc*32..+31.
// Element i = t*512 + lane*16 + j;  lane%16 = N;  j maps K per the 16-bit
// A/B VGPR striping (lanes 0-15: K {0..7,16..23}; lanes 16-31: {8..15,24..31}).
__global__ __launch_bounds__(256)
void prep_weights(const float* __restrict__ W, __bf16* __restrict__ dst,
                  int dout, int din) {
    int i = blockIdx.x * blockDim.x + threadIdx.x;
    int total = dout * din;
    if (i >= total) return;
    int t = i >> 9, r = i & 511;
    int lane = r >> 4, j = r & 15;
    int KC = din >> 5;
    int nt = t / KC, kc = t - nt * KC;
    int n = lane & 15;
    int v = j >> 1, odd = j & 1;
    const int tabLo[8] = {0, 2, 4, 6, 16, 18, 20, 22};
    const int tabHi[8] = {8, 10, 12, 14, 24, 26, 28, 30};
    int k = kc * 32 + ((lane < 16) ? tabLo[v] : tabHi[v]) + odd;
    dst[i] = __builtin_bit_cast(__bf16, bfbits(W[(size_t)(nt * 16 + n) * din + k]));
}

// ---------------------------------------------------------------------------
// Edge kernel: per wave = 16 edges. 4 waves/block, 32KB LDS per wave.
__global__ __launch_bounds__(128)
void edge_kernel(const float* __restrict__ x, const int* __restrict__ ei,
                 const float* __restrict__ ea, EdgeParams P,
                 float* __restrict__ aggT2D, float* __restrict__ aggD2T, int E) {
    extern __shared__ char smem[];
    const int lane = threadIdx.x & 31;
    const int wave = threadIdx.x >> 5;
    const int ebase = (blockIdx.x * 4 + wave) * 16;
    if (ebase >= E) return;

    char*   wb    = smem + wave * (32 * 1024);
    __bf16* bufIn = (__bf16*)wb;                 //  8KB: einput bf16 [16][256]
    __bf16* bufB  = (__bf16*)(wb + 8 * 1024);    //  8KB: LN1 output bf16 [16][256]
    float*  bufH  = (float*)(wb + 16 * 1024);    // 16KB: raw matmul f32 [16][256]

    const int row = lane & 15;     // M (and N for C-tile stores)
    const int hi  = lane >> 4;

    int rowv = 0, colv = 0;
    if (lane < 16) { rowv = ei[ebase + lane]; colv = ei[E + ebase + lane]; }

    // speculative prefetch of next tile's edge_attr rows -> global_prefetch_b8
    if (lane < 16 && (ebase + 16 + lane) < E)
        __builtin_prefetch(ea + (size_t)(ebase + 16 + lane) * EDIM, 0, 1);

    // ---- stage einput = [x[col] | edge_attr] as bf16 into LDS --------------
    {
        const int k0 = lane * 8;           // 32 lanes x 8 floats = 256
        for (int m = 0; m < 16; ++m) {
            int e  = ebase + m;
            int cm = __shfl(colv, m);
            const float* src = (k0 < XDIM) ? (x + (size_t)cm * XDIM + k0)
                                           : (ea + (size_t)e * EDIM + (k0 - XDIM));
            float4 f0 = ((const float4*)src)[0];
            float4 f1 = ((const float4*)src)[1];
            ((uint4*)(bufIn + m * (XDIM + EDIM)))[lane] =
                make_uint4(pk2(f0.x, f0.y), pk2(f0.z, f0.w),
                           pk2(f1.x, f1.y), pk2(f1.z, f1.w));
        }
    }
    __builtin_amdgcn_wave_barrier();

    // ---- cache the 8 A-chunks (K=256) in VGPRs (64 VGPRs) ------------------
    v16bf a[8];
#pragma unroll
    for (int kc = 0; kc < 8; ++kc) {
        const __bf16* rp = bufIn + row * 256 + kc * 32 + hi * 8;
        a[kc] = ldA16(rp, rp + 16);
    }

#pragma unroll 1
    for (int m = 0; m < 2; ++m) {
        const MlpP& L = P.mlp[m];

        // ---- layer 1: [16,256] @ [256,256]^T -------------------------------
#pragma unroll 1
        for (int nt = 0; nt < 16; ++nt) {
            v8f c = {};
#pragma unroll
            for (int kc = 0; kc < 8; ++kc) {
                v16bf b = ldB16(L.w0, nt * 8 + kc, lane);
                c = __builtin_amdgcn_wmma_f32_16x16x32_bf16(
                        false, a[kc], false, b, (short)0, c, false, false);
            }
            float bias = L.b0[nt * 16 + row];   // C layout: lane%16 == N
#pragma unroll
            for (int r = 0; r < 8; ++r)
                bufH[(hi * 8 + r) * 256 + nt * 16 + row] = c[r] + bias;
        }
        __builtin_amdgcn_wave_barrier();

        // ---- LayerNorm(256) + ReLU -> bf16 ---------------------------------
        {
            const float* hp = bufH + row * 256 + hi * 128;
            float s = 0.f, ss = 0.f;
#pragma unroll 4
            for (int k = 0; k < 128; ++k) { float v = hp[k]; s += v; ss += v * v; }
            s  += __shfl_xor(s, 16);
            ss += __shfl_xor(ss, 16);
            float mu  = s * (1.f / 256.f);
            float inv = rsqrtf(ss * (1.f / 256.f) - mu * mu + LN_EPS);
            const float* g  = L.g0  + hi * 128;
            const float* bb = L.bb0 + hi * 128;
            unsigned int* dst = (unsigned int*)(bufB + row * 256 + hi * 128);
#pragma unroll 2
            for (int k = 0; k < 128; k += 2) {
                float v0 = fmaxf((hp[k]     - mu) * inv * g[k]     + bb[k],     0.f);
                float v1 = fmaxf((hp[k + 1] - mu) * inv * g[k + 1] + bb[k + 1], 0.f);
                dst[k >> 1] = pk2(v0, v1);
            }
        }
        __builtin_amdgcn_wave_barrier();

        // ---- layer 2: [16,256] @ [256,128]^T -------------------------------
        v16bf a2[8];
#pragma unroll
        for (int kc = 0; kc < 8; ++kc) {
            const __bf16* rp = bufB + row * 256 + kc * 32 + hi * 8;
            a2[kc] = ldA16(rp, rp + 16);
        }
#pragma unroll 1
        for (int nt = 0; nt < 8; ++nt) {
            v8f c = {};
#pragma unroll
            for (int kc = 0; kc < 8; ++kc) {
                v16bf b = ldB16(L.w1, nt * 8 + kc, lane);
                c = __builtin_amdgcn_wmma_f32_16x16x32_bf16(
                        false, a2[kc], false, b, (short)0, c, false, false);
            }
            float bias = L.b1[nt * 16 + row];
#pragma unroll
            for (int r = 0; r < 8; ++r)
                bufH[(hi * 8 + r) * 128 + nt * 16 + row] = c[r] + bias;
        }
        __builtin_amdgcn_wave_barrier();

        // ---- LayerNorm(128) + ReLU + mask + atomic segment-sum -------------
        {
            const float* hp = bufH + row * 128 + hi * 64;
            float s = 0.f, ss = 0.f;
#pragma unroll 4
            for (int k = 0; k < 64; ++k) { float v = hp[k]; s += v; ss += v * v; }
            s  += __shfl_xor(s, 16);
            ss += __shfl_xor(ss, 16);
            float mu  = s * (1.f / 128.f);
            float inv = rsqrtf(ss * (1.f / 128.f) - mu * mu + LN_EPS);
            int re = __shfl(rowv, row);
            int ce = __shfl(colv, row);
            bool act = (m == 0) ? (re < ce) : (re > ce);
            float* agg = ((m == 0) ? aggD2T : aggT2D) + (size_t)re * 128 + hi * 64;
            const float* g  = L.g1  + hi * 64;
            const float* bb = L.bb1 + hi * 64;
            if (act) {
                for (int k = 0; k < 64; ++k) {
                    float v = fmaxf((hp[k] - mu) * inv * g[k] + bb[k], 0.f);
                    gatomic_add(agg + k, v);
                }
            }
        }
        __builtin_amdgcn_wave_barrier();
    }
}

// ---------------------------------------------------------------------------
// Node kernel: out = node_mlp([agg_t2d|agg_d2t]) + self_mlp(x); 16 nodes/wave.
__global__ __launch_bounds__(128)
void node_kernel(const float* __restrict__ x, const float* __restrict__ aggT2D,
                 const float* __restrict__ aggD2T, NodeParams P,
                 float* __restrict__ out, int N) {
    extern __shared__ char smem[];
    const int lane = threadIdx.x & 31;
    const int wave = threadIdx.x >> 5;
    const int base = (blockIdx.x * 4 + wave) * 16;
    if (base >= N) return;

    char*   wb   = smem + wave * (28 * 1024);
    __bf16* bufF = (__bf16*)wb;                  //  8KB flow bf16 [16][256]
    __bf16* bufX = (__bf16*)(wb + 8 * 1024);     //  4KB x bf16 [16][128]
    float*  bufH = (float*)(wb + 12 * 1024);     //  8KB node branch f32 [16][128]
    float*  bufS = (float*)(wb + 20 * 1024);     //  8KB self branch f32 [16][128]

    const int row = lane & 15;
    const int hi  = lane >> 4;

    // ---- stage flow_total = [agg_t2d | agg_d2t] and x as bf16 --------------
    {
        const int k0 = lane * 8;
        for (int m = 0; m < 16; ++m) {
            size_t nd = (size_t)(base + m);
            const float* src = (k0 < 128) ? (aggT2D + nd * 128 + k0)
                                          : (aggD2T + nd * 128 + (k0 - 128));
            float4 f0 = ((const float4*)src)[0];
            float4 f1 = ((const float4*)src)[1];
            ((uint4*)(bufF + m * 256))[lane] =
                make_uint4(pk2(f0.x, f0.y), pk2(f0.z, f0.w),
                           pk2(f1.x, f1.y), pk2(f1.z, f1.w));
        }
        const int k1 = lane * 4;
        for (int m = 0; m < 16; ++m) {
            float4 f = *(const float4*)(x + (size_t)(base + m) * 128 + k1);
            ((uint2*)(bufX + m * 128))[lane] = make_uint2(pk2(f.x, f.y), pk2(f.z, f.w));
        }
    }
    __builtin_amdgcn_wave_barrier();

    v16bf af[8];
#pragma unroll
    for (int kc = 0; kc < 8; ++kc) {
        const __bf16* rp = bufF + row * 256 + kc * 32 + hi * 8;
        af[kc] = ldA16(rp, rp + 16);
    }
    v16bf ax[4];
#pragma unroll
    for (int kc = 0; kc < 4; ++kc) {
        const __bf16* rp = bufX + row * 128 + kc * 32 + hi * 8;
        ax[kc] = ldA16(rp, rp + 16);
    }

    // ---- node branch: [16,256]@[256,128]^T ---------------------------------
#pragma unroll 1
    for (int nt = 0; nt < 8; ++nt) {
        v8f c = {};
#pragma unroll
        for (int kc = 0; kc < 8; ++kc) {
            v16bf b = ldB16(P.node.w0, nt * 8 + kc, lane);
            c = __builtin_amdgcn_wmma_f32_16x16x32_bf16(
                    false, af[kc], false, b, (short)0, c, false, false);
        }
        float bias = P.node.b0[nt * 16 + row];
#pragma unroll
        for (int r = 0; r < 8; ++r)
            bufH[(hi * 8 + r) * 128 + nt * 16 + row] = c[r] + bias;
    }
    // ---- self branch: [16,128]@[128,128]^T ---------------------------------
#pragma unroll 1
    for (int nt = 0; nt < 8; ++nt) {
        v8f c = {};
#pragma unroll
        for (int kc = 0; kc < 4; ++kc) {
            v16bf b = ldB16(P.self.w0, nt * 4 + kc, lane);
            c = __builtin_amdgcn_wmma_f32_16x16x32_bf16(
                    false, ax[kc], false, b, (short)0, c, false, false);
        }
        float bias = P.self.b0[nt * 16 + row];
#pragma unroll
        for (int r = 0; r < 8; ++r)
            bufS[(hi * 8 + r) * 128 + nt * 16 + row] = c[r] + bias;
    }
    __builtin_amdgcn_wave_barrier();

    // ---- LN+ReLU both branches (in place, lane-exclusive regions) ----------
    float* hp = bufH + row * 128 + hi * 64;
    float* sp = bufS + row * 128 + hi * 64;
    {
        float s = 0.f, ss = 0.f;
#pragma unroll 4
        for (int k = 0; k < 64; ++k) { float v = hp[k]; s += v; ss += v * v; }
        s  += __shfl_xor(s, 16);
        ss += __shfl_xor(ss, 16);
        float mu  = s * (1.f / 128.f);
        float inv = rsqrtf(ss * (1.f / 128.f) - mu * mu + LN_EPS);
        const float* g  = P.node.g0  + hi * 64;
        const float* bb = P.node.bb0 + hi * 64;
        for (int k = 0; k < 64; ++k)
            hp[k] = fmaxf((hp[k] - mu) * inv * g[k] + bb[k], 0.f);
    }
    {
        float s = 0.f, ss = 0.f;
#pragma unroll 4
        for (int k = 0; k < 64; ++k) { float v = sp[k]; s += v; ss += v * v; }
        s  += __shfl_xor(s, 16);
        ss += __shfl_xor(ss, 16);
        float mu  = s * (1.f / 128.f);
        float inv = rsqrtf(ss * (1.f / 128.f) - mu * mu + LN_EPS);
        const float* g  = P.self.g0  + hi * 64;
        const float* bb = P.self.bb0 + hi * 64;
        for (int k = 0; k < 64; ++k)
            sp[k] = fmaxf((sp[k] - mu) * inv * g[k] + bb[k], 0.f);
    }
    __builtin_amdgcn_wave_barrier();

    // ---- sum + store -------------------------------------------------------
    float* op = out + (size_t)(base + row) * 128 + hi * 64;
    for (int k = 0; k < 64; ++k) op[k] = hp[k] + sp[k];
}

// ---------------------------------------------------------------------------
extern "C" void kernel_launch(void* const* d_in, const int* in_sizes, int n_in,
                              void* d_out, int out_size, void* d_ws, size_t ws_size,
                              hipStream_t stream) {
    const float* x  = (const float*)d_in[0];
    const int*   ei = (const int*)d_in[1];
    const float* ea = (const float*)d_in[2];
    const int N = in_sizes[0] / XDIM;
    const int E = in_sizes[1] / 2;

    // workspace layout: [aggT2D f32 N*128][aggD2T f32 N*128][bf16 weights ~0.5MB]
    char*  ws     = (char*)d_ws;
    float* aggT2D = (float*)ws;
    float* aggD2T = aggT2D + (size_t)N * 128;
    __bf16* wbase = (__bf16*)(ws + (size_t)2 * N * 128 * sizeof(float));
    __bf16* wd2t0 = wbase;            // 256*256
    __bf16* wd2t1 = wbase + 65536;    // 128*256
    __bf16* wt2d0 = wbase + 98304;    // 256*256
    __bf16* wt2d1 = wbase + 163840;   // 128*256
    __bf16* wnode = wbase + 196608;   // 128*256
    __bf16* wself = wbase + 229376;   // 128*128

    // zero aggregation buffers (harness does not re-init between replays)
    {
        int n4 = N * 64;  // 2*N*128 / 4
        zero_kernel<<<(n4 + 255) / 256, 256, 0, stream>>>((float4*)ws, n4);
    }
    // convert + swizzle weights fp32 -> bf16 WMMA-B tiles
    prep_weights<<<(65536 + 255) / 256, 256, 0, stream>>>((const float*)d_in[3],  wd2t0, 256, 256);
    prep_weights<<<(32768 + 255) / 256, 256, 0, stream>>>((const float*)d_in[7],  wd2t1, 128, 256);
    prep_weights<<<(65536 + 255) / 256, 256, 0, stream>>>((const float*)d_in[11], wt2d0, 256, 256);
    prep_weights<<<(32768 + 255) / 256, 256, 0, stream>>>((const float*)d_in[15], wt2d1, 128, 256);
    prep_weights<<<(32768 + 255) / 256, 256, 0, stream>>>((const float*)d_in[19], wnode, 128, 256);
    prep_weights<<<(16384 + 255) / 256, 256, 0, stream>>>((const float*)d_in[23], wself, 128, 128);

    EdgeParams EP;
    EP.mlp[0] = { wd2t0, (const float*)d_in[4],  (const float*)d_in[5],  (const float*)d_in[6],
                  wd2t1, (const float*)d_in[8],  (const float*)d_in[9],  (const float*)d_in[10] };
    EP.mlp[1] = { wt2d0, (const float*)d_in[12], (const float*)d_in[13], (const float*)d_in[14],
                  wt2d1, (const float*)d_in[16], (const float*)d_in[17], (const float*)d_in[18] };

    {
        int tiles  = (E + 15) / 16;
        int blocks = (tiles + 3) / 4;
        edge_kernel<<<blocks, 128, 4 * 32 * 1024, stream>>>(x, ei, ea, EP,
                                                            aggT2D, aggD2T, E);
    }

    NodeParams NP;
    NP.node = { wnode, (const float*)d_in[20], (const float*)d_in[21], (const float*)d_in[22],
                nullptr, nullptr, nullptr, nullptr };
    NP.self = { wself, (const float*)d_in[24], (const float*)d_in[25], (const float*)d_in[26],
                nullptr, nullptr, nullptr, nullptr };
    {
        int tiles  = (N + 15) / 16;
        int blocks = (tiles + 3) / 4;
        node_kernel<<<blocks, 128, 4 * 28 * 1024, stream>>>(x, aggT2D, aggD2T, NP,
                                                            (float*)d_out, N);
    }
}
